// TemplateEncoder_84310208020836
// MI455X (gfx1250) — compile-verified, matrix operand order checked by reading, not run on hardware
//
#include <hip/hip_runtime.h>
#include <hip/hip_bf16.h>

typedef __attribute__((ext_vector_type(16))) _Float16 v16h;
typedef __attribute__((ext_vector_type(8)))  _Float16 v8h;
typedef __attribute__((ext_vector_type(8)))  float    v8f;

#define B_   16
#define N_   64
#define M_   256
#define L_   20
#define H_   128
#define TAU_ 64

// ---------- WMMA fragment loader (CDNA5 16-bit layout, CDNA5 ISA §7.12.2) ----------
// Works for A tiles (row-major [16][K]) and for B tiles stored K-transposed ([16 cols][K]).
// lane<16: row=lane, k = {0..7, 16..23}; lane>=16: row=lane-16, k = {8..15, 24..31}.
// Two contiguous 16-byte runs -> 2x ds_load_b128.
__device__ __forceinline__ v16h frag_ld(const _Float16* p0, int ld, int lane) {
  const _Float16* p = p0 + (lane & 15) * ld + ((lane < 16) ? 0 : 8);
  v8h lo = *(const v8h*)p;
  v8h hi = *(const v8h*)(p + 16);
  return __builtin_shufflevector(lo, hi, 0, 1, 2, 3, 4, 5, 6, 7,
                                 8, 9, 10, 11, 12, 13, 14, 15);
}

__device__ __forceinline__ v8f wmma_f16(v16h a, v16h b, v8f c) {
  return __builtin_amdgcn_wmma_f32_16x16x32_f16(false, a, false, b, (short)0, c,
                                                false, false);
}

// ---------- weight f32 -> f16 with K-transpose: src [K][Nc] -> dst [Nc][K] ----------
__global__ void k_conv_w_t(const float* __restrict__ src, _Float16* __restrict__ dst,
                           int K, int Nc) {
  int i = blockIdx.x * blockDim.x + threadIdx.x;
  if (i < K * Nc) {
    int k = i / Nc, n = i % Nc;
    dst[n * K + k] = (_Float16)src[i];
  }
}

// ---------- kernel 1: point MLP -> maxpool over L -> + embeddings ----------
__global__ void __launch_bounds__(128)
k_point_mlp(const float* __restrict__ poly, const float* __restrict__ W1,
            const float* __restrict__ b1, const _Float16* __restrict__ W2t,
            const float* __restrict__ b2,
            const int* __restrict__ mtype, const int* __restrict__ mtl,
            const int* __restrict__ mroute,
            const float* __restrict__ type_emb, const float* __restrict__ tl_emb,
            const float* __restrict__ route_emb,
            float* __restrict__ h_pool) {
  const int bm = blockIdx.x;
  const int tid = threadIdx.x, lane = tid & 31, wv = tid >> 5;
  __shared__ alignas(16) _Float16 sA[32][H_];   // 8 KB
  __shared__ alignas(16) _Float16 sW[H_ * H_];  // 32 KB (W2 transposed: [N][K])
  __shared__ float sO[32][H_];                  // 16 KB
  const float* pp = poly + (size_t)bm * (L_ * 2);
  __builtin_prefetch(W2t, 0, 0);
  {
    const v8h* ws = (const v8h*)W2t;
    v8h* wd = (v8h*)sW;
    for (int i = tid; i < (H_ * H_) / 8; i += 128) wd[i] = ws[i];
  }
  for (int i = tid; i < 32 * H_; i += 128) {
    int r = i >> 7, c = i & 127;
    float v = 0.f;
    if (r < L_) {
      float x = pp[r * 2], y = pp[r * 2 + 1];
      v = fmaxf(fmaf(x, W1[c], fmaf(y, W1[H_ + c], b1[c])), 0.f);
    }
    sA[r][c] = (_Float16)v;
  }
  __syncthreads();
#pragma unroll
  for (int tt = 0; tt < 4; ++tt) {
    int t = wv * 4 + tt, tr = t >> 3, tc = t & 7;
    v8f acc = {};
#pragma unroll
    for (int kk = 0; kk < 4; ++kk)
      acc = wmma_f16(frag_ld(&sA[tr * 16][kk * 32], H_, lane),
                     frag_ld(&sW[(tc * 16) * H_ + kk * 32], H_, lane), acc);
    int n = lane & 15, rb = (lane < 16) ? 0 : 8;
#pragma unroll
    for (int r = 0; r < 8; ++r) {
      int row = tr * 16 + rb + r, col = tc * 16 + n;
      sO[row][col] = fmaxf(acc[r] + b2[col], 0.f);
    }
  }
  __syncthreads();
  int ty = mtype[bm], tl = mtl[bm], rt = mroute[bm];
  for (int c = tid; c < H_; c += 128) {
    float mx = sO[0][c];
#pragma unroll
    for (int r = 1; r < L_; ++r) mx = fmaxf(mx, sO[r][c]);
    h_pool[(size_t)bm * H_ + c] =
        mx + type_emb[ty * H_ + c] + tl_emb[tl * H_ + c] + route_emb[rt * H_ + c];
  }
}

// ---------- kernel 2: map out MLP, masked; writes h_map TRANSPOSED f16 [B][H][M] ----------
__global__ void __launch_bounds__(128)
k_map_out(const float* __restrict__ h_pool, const _Float16* __restrict__ W1t,
          const float* __restrict__ b1, const _Float16* __restrict__ W2t,
          const float* __restrict__ b2, const unsigned char* __restrict__ mmask,
          _Float16* __restrict__ h_map_ht) {
  const int row0 = blockIdx.x * 32;
  const int tid = threadIdx.x, lane = tid & 31, wv = tid >> 5;
  __shared__ alignas(16) _Float16 sX[32][H_];
  __shared__ alignas(16) _Float16 sW[H_ * H_];
  __shared__ alignas(16) _Float16 sH[32][H_];
  for (int i = tid; i < 32 * H_; i += 128) {
    int r = i >> 7, c = i & 127;
    sX[r][c] = (_Float16)h_pool[(size_t)(row0 + r) * H_ + c];
  }
  {
    const v8h* ws = (const v8h*)W1t;
    v8h* wd = (v8h*)sW;
    for (int i = tid; i < (H_ * H_) / 8; i += 128) wd[i] = ws[i];
  }
  __syncthreads();
#pragma unroll
  for (int tt = 0; tt < 4; ++tt) {
    int t = wv * 4 + tt, tr = t >> 3, tc = t & 7;
    v8f acc = {};
#pragma unroll
    for (int kk = 0; kk < 4; ++kk)
      acc = wmma_f16(frag_ld(&sX[tr * 16][kk * 32], H_, lane),
                     frag_ld(&sW[(tc * 16) * H_ + kk * 32], H_, lane), acc);
    int n = lane & 15, rb = (lane < 16) ? 0 : 8;
#pragma unroll
    for (int r = 0; r < 8; ++r) {
      int row = tr * 16 + rb + r, col = tc * 16 + n;
      sH[row][col] = (_Float16)fmaxf(acc[r] + b1[col], 0.f);
    }
  }
  __syncthreads();
  {
    const v8h* ws = (const v8h*)W2t;
    v8h* wd = (v8h*)sW;
    for (int i = tid; i < (H_ * H_) / 8; i += 128) wd[i] = ws[i];
  }
  __syncthreads();
#pragma unroll
  for (int tt = 0; tt < 4; ++tt) {
    int t = wv * 4 + tt, tr = t >> 3, tc = t & 7;
    v8f acc = {};
#pragma unroll
    for (int kk = 0; kk < 4; ++kk)
      acc = wmma_f16(frag_ld(&sH[tr * 16][kk * 32], H_, lane),
                     frag_ld(&sW[(tc * 16) * H_ + kk * 32], H_, lane), acc);
    int n = lane & 15, rb = (lane < 16) ? 0 : 8;
#pragma unroll
    for (int r = 0; r < 8; ++r) {
      int grow = row0 + tr * 16 + rb + r, col = tc * 16 + n;
      int bb = grow >> 8, mm = grow & 255;  // M_ == 256
      float v = (acc[r] + b2[col]) * (mmask[grow] ? 1.f : 0.f);
      h_map_ht[((size_t)bb * H_ + col) * M_ + mm] = (_Float16)v;
    }
  }
}

// ---------- kernel 3: agent encoder; writes a_feat f32 [B*N][H] and transposed f16 [B][H][N] ----------
__global__ void __launch_bounds__(128)
k_agent(const float* __restrict__ st, const float* __restrict__ W1,
        const float* __restrict__ b1, const _Float16* __restrict__ W2t,
        const float* __restrict__ b2, const _Float16* __restrict__ W3t,
        const float* __restrict__ b3, float* __restrict__ a_feat,
        _Float16* __restrict__ a_feat_ht) {
  const int row0 = blockIdx.x * 32;
  const int tid = threadIdx.x, lane = tid & 31, wv = tid >> 5;
  __shared__ alignas(16) _Float16 sX[32][H_];
  __shared__ alignas(16) _Float16 sW[H_ * H_];
  __shared__ alignas(16) _Float16 sH[32][H_];
  for (int i = tid; i < 32 * H_; i += 128) {
    int r = i >> 7, c = i & 127;
    const float* s = st + (size_t)(row0 + r) * 5;
    float v = b1[c];
#pragma unroll
    for (int j = 0; j < 5; ++j) v = fmaf(s[j], W1[j * H_ + c], v);
    sX[r][c] = (_Float16)fmaxf(v, 0.f);
  }
  {
    const v8h* ws = (const v8h*)W2t;
    v8h* wd = (v8h*)sW;
    for (int i = tid; i < (H_ * H_) / 8; i += 128) wd[i] = ws[i];
  }
  __syncthreads();
#pragma unroll
  for (int tt = 0; tt < 4; ++tt) {
    int t = wv * 4 + tt, tr = t >> 3, tc = t & 7;
    v8f acc = {};
#pragma unroll
    for (int kk = 0; kk < 4; ++kk)
      acc = wmma_f16(frag_ld(&sX[tr * 16][kk * 32], H_, lane),
                     frag_ld(&sW[(tc * 16) * H_ + kk * 32], H_, lane), acc);
    int n = lane & 15, rb = (lane < 16) ? 0 : 8;
#pragma unroll
    for (int r = 0; r < 8; ++r) {
      int row = tr * 16 + rb + r, col = tc * 16 + n;
      sH[row][col] = (_Float16)fmaxf(acc[r] + b2[col], 0.f);
    }
  }
  __syncthreads();
  {
    const v8h* ws = (const v8h*)W3t;
    v8h* wd = (v8h*)sW;
    for (int i = tid; i < (H_ * H_) / 8; i += 128) wd[i] = ws[i];
  }
  __syncthreads();
#pragma unroll
  for (int tt = 0; tt < 4; ++tt) {
    int t = wv * 4 + tt, tr = t >> 3, tc = t & 7;
    v8f acc = {};
#pragma unroll
    for (int kk = 0; kk < 4; ++kk)
      acc = wmma_f16(frag_ld(&sH[tr * 16][kk * 32], H_, lane),
                     frag_ld(&sW[(tc * 16) * H_ + kk * 32], H_, lane), acc);
    int n = lane & 15, rb = (lane < 16) ? 0 : 8;
#pragma unroll
    for (int r = 0; r < 8; ++r) {
      int grow = row0 + tr * 16 + rb + r, col = tc * 16 + n;
      int bb = grow >> 6, nn2 = grow & 63;  // N_ == 64
      float v = acc[r] + b3[col];
      a_feat[(size_t)grow * H_ + col] = v;
      a_feat_ht[((size_t)bb * H_ + col) * N_ + nn2] = (_Float16)v;
    }
  }
}

// ---------- kernel 4: agent->map scores + masked softmax + WMMA attn@h_map ----------
__global__ void __launch_bounds__(128)
k_map_attn(const float* __restrict__ agents, const unsigned char* __restrict__ amask,
           const unsigned char* __restrict__ mmask, const float* __restrict__ poly,
           const float* __restrict__ mW1, const float* __restrict__ mb1,
           const float* __restrict__ mW2, const float* __restrict__ mb2,
           const _Float16* __restrict__ h_map_ht, float* __restrict__ map_ctx) {
  const int b = blockIdx.x >> 2, n0 = (blockIdx.x & 3) * 16;
  const int tid = threadIdx.x, lane = tid & 31, wv = tid >> 5;
  __shared__ float sLog[16][M_];                  // 16 KB
  __shared__ alignas(16) _Float16 sAttn[16][M_];  // 8 KB
  __shared__ alignas(16) _Float16 sBt[128][32];   // 8 KB (h_map^T K-chunk: [col][k])
  __shared__ float sCen[M_][2];
  __shared__ float sW1[3 * H_], sb1v[H_], sW2v[H_];
  for (int i = tid; i < 3 * H_; i += 128) sW1[i] = mW1[i];
  for (int i = tid; i < H_; i += 128) { sb1v[i] = mb1[i]; sW2v[i] = mW2[i]; }
  for (int m = tid; m < M_; m += 128) {
    const float* pp = poly + (size_t)(b * M_ + m) * (L_ * 2);
    float sx = 0.f, sy = 0.f;
#pragma unroll
    for (int l = 0; l < L_; ++l) { sx += pp[2 * l]; sy += pp[2 * l + 1]; }
    sCen[m][0] = sx * (1.f / L_);
    sCen[m][1] = sy * (1.f / L_);
  }
  __syncthreads();
  const float bias2 = mb2[0];
  for (int i = tid; i < 16 * M_; i += 128) {
    int r = i >> 8, m = i & 255;
    int n = n0 + r;
    const float* as = agents + (size_t)(b * N_ + n) * 5;
    float dx = sCen[m][0] - as[0], dy = sCen[m][1] - as[1];
    float cy = __cosf(as[2]), sy = __sinf(as[2]);
    float s = cy * dx + sy * dy, d = -sy * dx + cy * dy;
    float dist = sqrtf(s * s + d * d + 1e-9f);
    float acc = bias2;
    for (int h = 0; h < H_; ++h) {
      float hv = fmaf(s, sW1[h], fmaf(d, sW1[H_ + h], fmaf(dist, sW1[2 * H_ + h], sb1v[h])));
      acc = fmaf(fmaxf(hv, 0.f), sW2v[h], acc);
    }
    bool keep = mmask[b * M_ + m] && amask[b * N_ + n];
    sLog[r][m] = keep ? acc : -1e9f;
  }
  __syncthreads();
  if (tid < 16) {
    float mx = -3e38f;
    for (int m = 0; m < M_; ++m) mx = fmaxf(mx, sLog[tid][m]);
    float tot = 0.f;
    for (int m = 0; m < M_; ++m) {
      float lv = sLog[tid][m];
      float e = (lv <= -1e8f) ? 0.f : __expf(lv - mx);
      sLog[tid][m] = e;
      tot += e;
    }
    float inv = 1.f / fmaxf(tot, 1e-9f);
    for (int m = 0; m < M_; ++m) sAttn[tid][m] = (_Float16)(sLog[tid][m] * inv);
  }
  __syncthreads();
  const int tc0 = wv * 2, tc1 = wv * 2 + 1;
  v8f acc0 = {}, acc1 = {};
  for (int kk = 0; kk < 8; ++kk) {
    // stage 128 cols x 32 k of h_map^T as 16B vectors
    for (int i = tid; i < 512; i += 128) {
      int c = i >> 2, rblk = i & 3;
      *(v8h*)&sBt[c][rblk * 8] =
          *(const v8h*)&h_map_ht[((size_t)b * H_ + c) * M_ + kk * 32 + rblk * 8];
    }
    __syncthreads();
    v16h fa = frag_ld(&sAttn[0][kk * 32], M_, lane);
    acc0 = wmma_f16(fa, frag_ld(&sBt[tc0 * 16][0], 32, lane), acc0);
    acc1 = wmma_f16(fa, frag_ld(&sBt[tc1 * 16][0], 32, lane), acc1);
    __syncthreads();
  }
  int nn = lane & 15, rb = (lane < 16) ? 0 : 8;
#pragma unroll
  for (int r = 0; r < 8; ++r) {
    int gn = n0 + rb + r;
    size_t base = (size_t)(b * N_ + gn) * H_;
    map_ctx[base + tc0 * 16 + nn] = acc0[r];
    map_ctx[base + tc1 * 16 + nn] = acc1[r];
  }
}

// ---------- kernel 5: neighbor scores + masked softmax + WMMA attn@a ----------
__global__ void __launch_bounds__(128)
k_nbr_attn(const float* __restrict__ agents, const unsigned char* __restrict__ amask,
           const float* __restrict__ nW1, const float* __restrict__ nb1,
           const float* __restrict__ nW2, const float* __restrict__ nb2,
           const _Float16* __restrict__ a_feat_ht, float* __restrict__ nbr_ctx) {
  const int b = blockIdx.x >> 2, n0 = (blockIdx.x & 3) * 16;
  const int tid = threadIdx.x, lane = tid & 31, wv = tid >> 5;
  __shared__ float sLog[16][N_];
  __shared__ alignas(16) _Float16 sAttn[16][N_];
  __shared__ alignas(16) _Float16 sBt[128][N_];  // 16 KB (a_feat^T: [col][n])
  __shared__ float sW1[6 * H_], sb1v[H_], sW2v[H_];
  for (int i = tid; i < 6 * H_; i += 128) sW1[i] = nW1[i];
  for (int i = tid; i < H_; i += 128) { sb1v[i] = nb1[i]; sW2v[i] = nW2[i]; }
  for (int i = tid; i < (H_ * N_) / 8; i += 128)
    ((v8h*)&sBt[0][0])[i] = ((const v8h*)&a_feat_ht[(size_t)b * H_ * N_])[i];
  __syncthreads();
  const float bias2 = nb2[0];
  for (int i = tid; i < 16 * N_; i += 128) {
    int r = i >> 6, j = i & 63;
    int n = n0 + r;
    const float* ai = agents + (size_t)(b * N_ + n) * 5;
    const float* aj = agents + (size_t)(b * N_ + j) * 5;
    float dx = aj[0] - ai[0], dy = aj[1] - ai[1];
    float cy = __cosf(ai[2]), sy = __sinf(ai[2]);
    float ns = cy * dx + sy * dy, nd = -sy * dx + cy * dy;
    float ndist = sqrtf(ns * ns + nd * nd + 1e-9f);
    float dyaw = aj[2] - ai[2];
    float sdy = __sinf(dyaw), cdy = __cosf(dyaw);
    float dspd = aj[3] - ai[3];
    float acc = bias2;
    for (int h = 0; h < H_; ++h) {
      float hv = ns * sW1[h] + nd * sW1[H_ + h] + ndist * sW1[2 * H_ + h] +
                 sdy * sW1[3 * H_ + h] + cdy * sW1[4 * H_ + h] +
                 dspd * sW1[5 * H_ + h] + sb1v[h];
      acc = fmaf(fmaxf(hv, 0.f), sW2v[h], acc);
    }
    bool keep = amask[b * N_ + j] && amask[b * N_ + n] && (j != n) && (ndist <= 60.0f);
    sLog[r][j] = keep ? acc : -1e9f;
  }
  __syncthreads();
  if (tid < 16) {
    float mx = -3e38f;
    for (int j = 0; j < N_; ++j) mx = fmaxf(mx, sLog[tid][j]);
    float tot = 0.f;
    for (int j = 0; j < N_; ++j) {
      float lv = sLog[tid][j];
      float e = (lv <= -1e8f) ? 0.f : __expf(lv - mx);
      sLog[tid][j] = e;
      tot += e;
    }
    float inv = 1.f / fmaxf(tot, 1e-9f);
    for (int j = 0; j < N_; ++j) sAttn[tid][j] = (_Float16)(sLog[tid][j] * inv);
  }
  __syncthreads();
  const int tc0 = wv * 2, tc1 = wv * 2 + 1;
  v8f acc0 = {}, acc1 = {};
#pragma unroll
  for (int kk = 0; kk < 2; ++kk) {
    v16h fa = frag_ld(&sAttn[0][kk * 32], N_, lane);
    acc0 = wmma_f16(fa, frag_ld(&sBt[tc0 * 16][kk * 32], N_, lane), acc0);
    acc1 = wmma_f16(fa, frag_ld(&sBt[tc1 * 16][kk * 32], N_, lane), acc1);
  }
  int nn = lane & 15, rb = (lane < 16) ? 0 : 8;
#pragma unroll
  for (int r = 0; r < 8; ++r) {
    int gn = n0 + rb + r;
    size_t base = (size_t)(b * N_ + gn) * H_;
    nbr_ctx[base + tc0 * 16 + nn] = acc0[r];
    nbr_ctx[base + tc1 * 16 + nn] = acc1[r];
  }
}

// ---------- kernel 6: tau head ----------
__global__ void __launch_bounds__(256)
k_tau(const float* __restrict__ a_feat, const float* __restrict__ map_ctx,
      const float* __restrict__ nbr_ctx, const _Float16* __restrict__ W1t,
      const float* __restrict__ b1, const _Float16* __restrict__ W2t,
      const float* __restrict__ b2, const unsigned char* __restrict__ amask,
      float* __restrict__ out) {
  const int row0 = blockIdx.x * 32;
  const int tid = threadIdx.x, lane = tid & 31, wv = tid >> 5;
  __shared__ alignas(16) _Float16 sZ[32 * 384];  // 24 KB
  __shared__ alignas(16) _Float16 sWf[8192];     // 16 KB
  __shared__ alignas(16) _Float16 sH[32 * H_];   // 8 KB
  for (int i = tid; i < 32 * 384; i += 256) {
    int r = i / 384, c = i % 384;
    size_t gr = (size_t)(row0 + r) * H_;
    float v = (c < 128) ? a_feat[gr + c]
            : (c < 256) ? map_ctx[gr + c - 128]
                        : nbr_ctx[gr + c - 256];
    sZ[i] = (_Float16)v;
  }
  __syncthreads();
  const int t0 = wv * 2, t1 = wv * 2 + 1;
  const int tr = t0 >> 3, tc0 = t0 & 7, tc1 = t1 & 7;
  v8f acc0 = {}, acc1 = {};
  for (int kk = 0; kk < 12; ++kk) {
    // stage W1^T chunk: 128 out-cols x 32 k -> sWf[c*32 + k]
    for (int i = tid; i < 512; i += 256) {
      int c = i >> 2, rblk = i & 3;
      *(v8h*)&sWf[c * 32 + rblk * 8] =
          *(const v8h*)&W1t[(size_t)c * 384 + kk * 32 + rblk * 8];
    }
    __syncthreads();
    v16h fa = frag_ld(&sZ[tr * 16 * 384 + kk * 32], 384, lane);
    acc0 = wmma_f16(fa, frag_ld(&sWf[(tc0 * 16) * 32], 32, lane), acc0);
    acc1 = wmma_f16(fa, frag_ld(&sWf[(tc1 * 16) * 32], 32, lane), acc1);
    __syncthreads();
  }
  int nn = lane & 15, rb = (lane < 16) ? 0 : 8;
#pragma unroll
  for (int r = 0; r < 8; ++r) {
    int row = tr * 16 + rb + r;
    int c0 = tc0 * 16 + nn, c1 = tc1 * 16 + nn;
    sH[row * H_ + c0] = (_Float16)fmaxf(acc0[r] + b1[c0], 0.f);
    sH[row * H_ + c1] = (_Float16)fmaxf(acc1[r] + b1[c1], 0.f);
  }
  __syncthreads();
  {
    const v8h* ws = (const v8h*)W2t;  // [TAU][H] transposed
    v8h* wd = (v8h*)sWf;
    for (int i = tid; i < (H_ * TAU_) / 8; i += 256) wd[i] = ws[i];
  }
  __syncthreads();
  const int tr2 = wv >> 2, tc2 = wv & 3;
  v8f acc = {};
#pragma unroll
  for (int kk = 0; kk < 4; ++kk)
    acc = wmma_f16(frag_ld(&sH[tr2 * 16 * H_ + kk * 32], H_, lane),
                   frag_ld(&sWf[(tc2 * 16) * H_ + kk * 32], H_, lane), acc);
#pragma unroll
  for (int r = 0; r < 8; ++r) {
    int row = row0 + tr2 * 16 + rb + r;
    int col = tc2 * 16 + nn;
    float v = (acc[r] + b2[col]) * (amask[row] ? 1.f : 0.f);
    out[(size_t)row * TAU_ + col] = v;
  }
}

// ---------- host ----------
extern "C" void kernel_launch(void* const* d_in, const int* in_sizes, int n_in,
                              void* d_out, int out_size, void* d_ws, size_t ws_size,
                              hipStream_t stream) {
  (void)in_sizes; (void)n_in; (void)out_size; (void)ws_size;
  const float*         agents = (const float*)d_in[0];
  const unsigned char* amask  = (const unsigned char*)d_in[1];
  const float*         poly   = (const float*)d_in[2];
  const unsigned char* mmask  = (const unsigned char*)d_in[3];
  const int*           mtype  = (const int*)d_in[4];
  const int*           mtl    = (const int*)d_in[5];
  const int*           mroute = (const int*)d_in[6];
  int pi = 7;
  const float* point_W1 = (const float*)d_in[pi++];
  const float* point_b1 = (const float*)d_in[pi++];
  const float* point_W2 = (const float*)d_in[pi++];
  const float* point_b2 = (const float*)d_in[pi++];
  const float* type_emb = (const float*)d_in[pi++];
  const float* tl_emb   = (const float*)d_in[pi++];
  const float* route_emb= (const float*)d_in[pi++];
  const float* mout_W1  = (const float*)d_in[pi++];
  const float* mout_b1  = (const float*)d_in[pi++];
  const float* mout_W2  = (const float*)d_in[pi++];
  const float* mout_b2  = (const float*)d_in[pi++];
  const float* ag_W1    = (const float*)d_in[pi++];
  const float* ag_b1    = (const float*)d_in[pi++];
  const float* ag_W2    = (const float*)d_in[pi++];
  const float* ag_b2    = (const float*)d_in[pi++];
  const float* ag_W3    = (const float*)d_in[pi++];
  const float* ag_b3    = (const float*)d_in[pi++];
  const float* mrel_W1  = (const float*)d_in[pi++];
  const float* mrel_b1  = (const float*)d_in[pi++];
  const float* mrel_W2  = (const float*)d_in[pi++];
  const float* mrel_b2  = (const float*)d_in[pi++];
  const float* nrel_W1  = (const float*)d_in[pi++];
  const float* nrel_b1  = (const float*)d_in[pi++];
  const float* nrel_W2  = (const float*)d_in[pi++];
  const float* nrel_b2  = (const float*)d_in[pi++];
  const float* tau_W1   = (const float*)d_in[pi++];
  const float* tau_b1   = (const float*)d_in[pi++];
  const float* tau_W2   = (const float*)d_in[pi++];
  const float* tau_b2   = (const float*)d_in[pi++];

  char* wp = (char*)d_ws;
  auto carve = [&](size_t bytes) -> char* {
    char* p = wp;
    wp += (bytes + 255) & ~(size_t)255;
    return p;
  };
  float*    h_pool     = (float*)carve((size_t)B_ * M_ * H_ * 4);
  _Float16* h_map_ht   = (_Float16*)carve((size_t)B_ * H_ * M_ * 2);
  float*    a_feat     = (float*)carve((size_t)B_ * N_ * H_ * 4);
  _Float16* a_feat_ht  = (_Float16*)carve((size_t)B_ * H_ * N_ * 2);
  float*    map_ctx    = (float*)carve((size_t)B_ * N_ * H_ * 4);
  float*    nbr_ctx    = (float*)carve((size_t)B_ * N_ * H_ * 4);
  _Float16* point_W2t  = (_Float16*)carve(H_ * H_ * 2);
  _Float16* mout_W1t   = (_Float16*)carve(H_ * H_ * 2);
  _Float16* mout_W2t   = (_Float16*)carve(H_ * H_ * 2);
  _Float16* ag_W2t     = (_Float16*)carve(H_ * H_ * 2);
  _Float16* ag_W3t     = (_Float16*)carve(H_ * H_ * 2);
  _Float16* tau_W1t    = (_Float16*)carve(384 * H_ * 2);
  _Float16* tau_W2t    = (_Float16*)carve(H_ * TAU_ * 2);

  k_conv_w_t<<<(H_ * H_ + 255) / 256, 256, 0, stream>>>(point_W2, point_W2t, H_, H_);
  k_conv_w_t<<<(H_ * H_ + 255) / 256, 256, 0, stream>>>(mout_W1, mout_W1t, H_, H_);
  k_conv_w_t<<<(H_ * H_ + 255) / 256, 256, 0, stream>>>(mout_W2, mout_W2t, H_, H_);
  k_conv_w_t<<<(H_ * H_ + 255) / 256, 256, 0, stream>>>(ag_W2, ag_W2t, H_, H_);
  k_conv_w_t<<<(H_ * H_ + 255) / 256, 256, 0, stream>>>(ag_W3, ag_W3t, H_, H_);
  k_conv_w_t<<<(384 * H_ + 255) / 256, 256, 0, stream>>>(tau_W1, tau_W1t, 384, H_);
  k_conv_w_t<<<(H_ * TAU_ + 255) / 256, 256, 0, stream>>>(tau_W2, tau_W2t, H_, TAU_);

  k_point_mlp<<<B_ * M_, 128, 0, stream>>>(poly, point_W1, point_b1, point_W2t,
                                           point_b2, mtype, mtl, mroute, type_emb,
                                           tl_emb, route_emb, h_pool);
  k_map_out<<<(B_ * M_) / 32, 128, 0, stream>>>(h_pool, mout_W1t, mout_b1, mout_W2t,
                                                mout_b2, mmask, h_map_ht);
  k_agent<<<(B_ * N_) / 32, 128, 0, stream>>>(agents, ag_W1, ag_b1, ag_W2t, ag_b2,
                                              ag_W3t, ag_b3, a_feat, a_feat_ht);
  k_map_attn<<<B_ * 4, 128, 0, stream>>>(agents, amask, mmask, poly, mrel_W1, mrel_b1,
                                         mrel_W2, mrel_b2, h_map_ht, map_ctx);
  k_nbr_attn<<<B_ * 4, 128, 0, stream>>>(agents, amask, nrel_W1, nrel_b1, nrel_W2,
                                         nrel_b2, a_feat_ht, nbr_ctx);
  k_tau<<<(B_ * N_) / 32, 256, 0, stream>>>(a_feat, map_ctx, nbr_ctx, tau_W1t, tau_b1,
                                            tau_W2t, tau_b2, amask, (float*)d_out);
}
